// LSTMSentimentAnalysis_71004399337676
// MI455X (gfx1250) — matrix operand order repaired and checked by reading, not compiled, not run on hardware
//
#include <hip/hip_runtime.h>

// ---------------------------------------------------------------------------
// CDNA5 (gfx1250) bf16 WMMA LSTM stack.
// All matmuls via v_wmma_f32_16x16x32_bf16 (wave32).
// ---------------------------------------------------------------------------

typedef __attribute__((ext_vector_type(16))) __bf16       v16bf;
typedef __attribute__((ext_vector_type(8)))  float        v8f;
typedef __attribute__((ext_vector_type(4)))  unsigned int u32x4;
typedef __attribute__((ext_vector_type(2)))  unsigned int u32x2;

union Frag {
  v16bf v;
  u32x4 q[2];
};

__device__ __forceinline__ unsigned short f2bf(float f) {
  unsigned int u = __float_as_uint(f);
  u += 0x7fffu + ((u >> 16) & 1u);          // round-to-nearest-even
  return (unsigned short)(u >> 16);
}
__device__ __forceinline__ float bf2f(unsigned short h) {
  return __uint_as_float(((unsigned int)h) << 16);
}
__device__ __forceinline__ float sigf(float x) { return 1.0f / (1.0f + __expf(-x)); }

__device__ __forceinline__ v8f wmma_bf16(Frag a, Frag b, v8f c) {
  // 8 args: (neg_a, A, neg_b, B, c_mod, C, reuse_a, reuse_b)
  return __builtin_amdgcn_wmma_f32_16x16x32_bf16(false, a.v, false, b.v,
                                                 (short)0, c, false, false);
}

// ---------------------------------------------------------------------------
// fp32 -> bf16 weight conversion
// ---------------------------------------------------------------------------
__global__ void k_cvt(const float* __restrict__ s, unsigned short* __restrict__ d, int n) {
  int i = blockIdx.x * 256 + threadIdx.x;
  if (i < n) d[i] = f2bf(s[i]);
}

// ---------------------------------------------------------------------------
// Word embedding gather + renorm (256 dims/row) -> X0 cols [0,256)
// grid = 16384 rows, block = 256
// ---------------------------------------------------------------------------
__global__ void k_embed_words(const int* __restrict__ words,
                              const float* __restrict__ emb,
                              unsigned short* __restrict__ X0) {
  int row = blockIdx.x;
  int tid = threadIdx.x;
  int idx = words[row];
  float v = emb[idx * 256 + tid];
  float ss = v * v;
#pragma unroll
  for (int o = 16; o > 0; o >>= 1) ss += __shfl_xor(ss, o, 32);
  __shared__ float wsum[8];
  if ((tid & 31) == 0) wsum[tid >> 5] = ss;
  __syncthreads();
  float tot = 0.f;
#pragma unroll
  for (int i = 0; i < 8; ++i) tot += wsum[i];
  float n = sqrtf(tot);
  float sc = (n > 1.0f) ? (1.0f / n) : 1.0f;  // max(n,1e-12)==n when n>1
  X0[row * 320 + tid] = f2bf(v * sc);
}

// ---------------------------------------------------------------------------
// Char embedding gather + renorm (64 dims/row), 4 rows per 256-thread block
// grid = 262144/4
// ---------------------------------------------------------------------------
__global__ void k_embed_chars(const int* __restrict__ chars,
                              const float* __restrict__ emb,
                              unsigned short* __restrict__ CE) {
  int tid = threadIdx.x;
  int r = blockIdx.x * 4 + (tid >> 6);
  int j = tid & 63;
  int idx = chars[r];
  float v = emb[idx * 64 + j];
  float ss = v * v;
#pragma unroll
  for (int o = 16; o > 0; o >>= 1) ss += __shfl_xor(ss, o, 32);
  __shared__ float wsum[8];
  if ((tid & 31) == 0) wsum[tid >> 5] = ss;
  __syncthreads();
  int rr = tid >> 6;
  float tot = wsum[rr * 2] + wsum[rr * 2 + 1];
  float n = sqrtf(tot);
  float sc = (n > 1.0f) ? (1.0f / n) : 1.0f;
  CE[r * 64 + j] = f2bf(v * sc);
}

// ---------------------------------------------------------------------------
// Char LSTM, fused input+recurrent WMMA, branch-free split K loops:
//   loop 1: K=[0,64)  A from CE (global),  B from wih
//   loop 2: K=[0,64)  A from h_{t-1} (LDS), B from whh
// grid = 1024 (16 samples each), block = 128 (4 waves; wave = 16 hidden units)
// Writes final h into X0 cols [256,320).
// ---------------------------------------------------------------------------
__global__ void k_char_lstm(const unsigned short* __restrict__ CE,
                            const unsigned short* __restrict__ wih,
                            const unsigned short* __restrict__ whh,
                            const float* __restrict__ bias,
                            unsigned short* __restrict__ X0) {
  int tid = threadIdx.x;
  int lane = tid & 31, hg = tid >> 5;         // hg: 0..3
  int sBase = blockIdx.x * 16;
  __shared__ __align__(16) unsigned short hb[2][16 * 72];  // padded stride 72
  for (int i = tid; i < 16 * 72; i += 128) hb[0][i] = 0;
  __syncthreads();
  int hcol = hg * 16 + (lane & 15);
  float bb[4];
#pragma unroll
  for (int g = 0; g < 4; ++g) bb[g] = bias[g * 64 + hcol];
  float cst[8];
#pragma unroll
  for (int e = 0; e < 8; ++e) cst[e] = 0.f;

  for (int t = 0; t < 16; ++t) {
    v8f acc[4];
#pragma unroll
    for (int g = 0; g < 4; ++g)
#pragma unroll
      for (int e = 0; e < 8; ++e) acc[g][e] = bb[g];

#pragma unroll 1  // rolled: keep wih loads in-loop (no hoist->spill)
    for (int kk = 0; kk < 2; ++kk) {  // input projection: A from CE
      Frag a;
      int r = (sBase + (lane & 15)) * 16 + t;
      const unsigned short* ap = CE + r * 64 + kk * 32 + (lane >> 4) * 16;
      a.q[0] = *(const u32x4*)ap;
      a.q[1] = *(const u32x4*)(ap + 8);
#pragma unroll
      for (int g = 0; g < 4; ++g) {
        Frag b;
        const unsigned short* wp =
            wih + (g * 64 + hcol) * 64 + kk * 32 + (lane >> 4) * 16;
        b.q[0] = *(const u32x4*)wp;
        b.q[1] = *(const u32x4*)(wp + 8);
        acc[g] = wmma_bf16(a, b, acc[g]);
      }
    }
#pragma unroll 1  // rolled: keep whh loads in-loop (no hoist->spill)
    for (int kk = 0; kk < 2; ++kk) {  // recurrent part: A from LDS h
      Frag a;
      const unsigned short* ap =
          &hb[t & 1][(lane & 15) * 72 + kk * 32 + (lane >> 4) * 16];
      a.q[0] = *(const u32x4*)ap;
      a.q[1] = *(const u32x4*)(ap + 8);
#pragma unroll
      for (int g = 0; g < 4; ++g) {
        Frag b;
        const unsigned short* wp =
            whh + (g * 64 + hcol) * 64 + kk * 32 + (lane >> 4) * 16;
        b.q[0] = *(const u32x4*)wp;
        b.q[1] = *(const u32x4*)(wp + 8);
        acc[g] = wmma_bf16(a, b, acc[g]);
      }
    }
#pragma unroll
    for (int e = 0; e < 8; ++e) {
      float cv = sigf(acc[1][e]) * cst[e] + sigf(acc[0][e]) * tanhf(acc[2][e]);
      cst[e] = cv;
      float hv = sigf(acc[3][e]) * tanhf(cv);
      unsigned short h16 = f2bf(hv);
      hb[(t + 1) & 1][((lane >> 4) * 8 + e) * 72 + hcol] = h16;
      if (t == 15) X0[(sBase + (lane >> 4) * 8 + e) * 320 + 256 + hcol] = h16;
    }
    __syncthreads();
  }
}

// ---------------------------------------------------------------------------
// GEMM for gate pre-activations:
//   pre = A(MxK,bf16) @ W(2048xK,bf16)^T + bias(2048,f32)
// Output layout is GATE-INTERLEAVED for the recurrence's packed loads:
//   pre[row][h][g]  (g=0..3 contiguous), i.e. index = row*2048 + (col&511)*4
//   + (col>>9).
// WG tile 128x128, 8 waves (wave = 16-row stripe, 8 col tiles), B staged in LDS
// ---------------------------------------------------------------------------
__global__ void k_gemm_pre(const unsigned short* __restrict__ A,
                           const unsigned short* __restrict__ W,
                           const float* __restrict__ bias,
                           unsigned short* __restrict__ out,
                           int M, int K) {
  int mBase = blockIdx.x * 128;
  int nBase = blockIdx.y * 128;
  int tid = threadIdx.x;
  int lane = tid & 31, mt = tid >> 5;
  __shared__ __align__(16) unsigned short Bt[128 * 40];  // padded stride 40
  v8f acc[8];
#pragma unroll
  for (int i = 0; i < 8; ++i)
#pragma unroll
    for (int e = 0; e < 8; ++e) acc[i][e] = 0.f;

  int nk = K >> 5;
  for (int kk = 0; kk < nk; ++kk) {
    __syncthreads();
    {  // cooperative stage of B 32x128 K-slab into LDS
      int n = tid >> 1, hf = tid & 1;
      const u32x4* src = (const u32x4*)(W + (nBase + n) * K + kk * 32 + hf * 16);
      u32x4* dst = (u32x4*)(Bt + n * 40 + hf * 16);
      dst[0] = src[0];
      dst[1] = src[1];
    }
    __syncthreads();
    Frag a;
    {
      int m = mBase + mt * 16 + (lane & 15);
      int k0 = kk * 32 + (lane >> 4) * 16;
      a.q[0] = *(const u32x4*)(A + m * K + k0);
      a.q[1] = *(const u32x4*)(A + m * K + k0 + 8);
    }
#pragma unroll
    for (int nt = 0; nt < 8; ++nt) {
      Frag b;
      const unsigned short* bp = Bt + (nt * 16 + (lane & 15)) * 40 + (lane >> 4) * 16;
      b.q[0] = *(const u32x4*)bp;
      b.q[1] = *(const u32x4*)(bp + 8);
      acc[nt] = wmma_bf16(a, b, acc[nt]);
    }
  }
#pragma unroll
  for (int nt = 0; nt < 8; ++nt) {
    int col = nBase + nt * 16 + (lane & 15);  // 0..2047 gate-major column
    int gi = (col >> 9);                      // gate 0..3
    int hi = (col & 511);                     // hidden unit
    float bv = bias[col];
#pragma unroll
    for (int e = 0; e < 8; ++e) {
      int row = mBase + mt * 16 + (lane >> 4) * 8 + e;
      out[row * 2048 + hi * 4 + gi] = f2bf(acc[nt][e] + bv);
    }
  }
}

// ---------------------------------------------------------------------------
// Main BiLSTM recurrence. grid = 4: blockIdx = dir*2 + batch-half.
// block = 1024 (32 waves). Wave owns 16 hidden units x 4 gates x 2 row-tiles
// (32 samples/WG, ~120 VGPRs/lane, no spills).
// h in single LDS buffer (32 x 520 bf16 padded, conflict-free b128 A-frags);
// c stays in VGPRs. K loop kept rolled so whh loads stay in-loop (L2 hits).
// pre* are gate-interleaved bf16 [row][h][g] -> one b64 load per 4 gates.
// Hout bf16 (16384 x 1024) [fwd | bwd].
// ---------------------------------------------------------------------------
__global__ void k_lstm_rec(const unsigned short* __restrict__ preF,
                           const unsigned short* __restrict__ preB,
                           const unsigned short* __restrict__ whhF,
                           const unsigned short* __restrict__ whhB,
                           unsigned short* __restrict__ Hout) {
  int dir = blockIdx.x >> 1;
  int half = blockIdx.x & 1;
  int sBase = half * 32;
  int tid = threadIdx.x;
  int lane = tid & 31, hg = tid >> 5;  // hg: 0..31
  const unsigned short* pre = dir ? preB : preF;
  const unsigned short* whh = dir ? whhB : whhF;
  __shared__ __align__(16) unsigned short hb[32 * 520];
  for (int i = tid; i < 32 * 520; i += 1024) hb[i] = 0;
  __syncthreads();
  float cst[2][8];
#pragma unroll
  for (int r = 0; r < 2; ++r)
#pragma unroll
    for (int e = 0; e < 8; ++e) cst[r][e] = 0.f;
  int hcol = hg * 16 + (lane & 15);

  for (int t = 0; t < 256; ++t) {
    int tt = dir ? (255 - t) : t;
    v8f acc[2][4];
#pragma unroll
    for (int r = 0; r < 2; ++r)
#pragma unroll
      for (int g = 0; g < 4; ++g)
#pragma unroll
        for (int e = 0; e < 8; ++e) acc[r][g][e] = 0.f;

#pragma unroll 1  // rolled: keep whh loads inside the loop (no hoist/spill)
    for (int kk = 0; kk < 16; ++kk) {
      Frag a0, a1;  // h_{t-1} tiles for the two row-tiles (LDS)
      const unsigned short* ap0 =
          &hb[(lane & 15) * 520 + kk * 32 + (lane >> 4) * 16];
      a0.q[0] = *(const u32x4*)ap0;
      a0.q[1] = *(const u32x4*)(ap0 + 8);
      const unsigned short* ap1 = ap0 + 16 * 520;
      a1.q[0] = *(const u32x4*)ap1;
      a1.q[1] = *(const u32x4*)(ap1 + 8);
#pragma unroll
      for (int g = 0; g < 4; ++g) {
        Frag b;  // whh rows are K-contiguous -> direct b128 loads (L2 resident)
        const unsigned short* wp =
            whh + (g * 512 + hcol) * 512 + kk * 32 + (lane >> 4) * 16;
        b.q[0] = *(const u32x4*)wp;
        b.q[1] = *(const u32x4*)(wp + 8);
        acc[0][g] = wmma_bf16(a0, b, acc[0][g]);
        acc[1][g] = wmma_bf16(a1, b, acc[1][g]);
      }
    }
    __syncthreads();  // all reads of h_{t-1} done before overwrite
#pragma unroll
    for (int r = 0; r < 2; ++r)
#pragma unroll
      for (int e = 0; e < 8; ++e) {
        int rowl = r * 16 + (lane >> 4) * 8 + e;
        int s = sBase + rowl;
        int prow = s * 256 + tt;
        u32x2 pk = *(const u32x2*)(pre + (prow * 512 + hcol) * 4);  // b64: 4 gates
        float gi = acc[r][0][e] + bf2f((unsigned short)(pk.x & 0xffffu));
        float gf = acc[r][1][e] + bf2f((unsigned short)(pk.x >> 16));
        float gg = acc[r][2][e] + bf2f((unsigned short)(pk.y & 0xffffu));
        float go = acc[r][3][e] + bf2f((unsigned short)(pk.y >> 16));
        float cv = sigf(gf) * cst[r][e] + sigf(gi) * tanhf(gg);
        cst[r][e] = cv;
        float hv = sigf(go) * tanhf(cv);
        unsigned short h16 = f2bf(hv);
        hb[rowl * 520 + hcol] = h16;
        Hout[prow * 1024 + dir * 512 + hcol] = h16;
      }
    __syncthreads();  // h_t visible to all waves before next step reads
  }
}

// ---------------------------------------------------------------------------
// Classifier: feat(64x1024) @ Wc(3x1024)^T + bc  (tiny: plain dot products)
// feat = [H2[b,255,0:512] | H2[b,0,512:1024]]
// ---------------------------------------------------------------------------
__global__ void k_classifier(const unsigned short* __restrict__ H2,
                             const float* __restrict__ Wc,
                             const float* __restrict__ bc,
                             float* __restrict__ out) {
  int tid = threadIdx.x;  // 192
  int b = tid / 3, o = tid % 3;
  float acc = bc[o];
  for (int k = 0; k < 512; ++k)
    acc += bf2f(H2[(b * 256 + 255) * 1024 + k]) * Wc[o * 1024 + k];
  for (int k = 512; k < 1024; ++k)
    acc += bf2f(H2[(b * 256 + 0) * 1024 + k]) * Wc[o * 1024 + k];
  out[b * 3 + o] = acc;
}

// ---------------------------------------------------------------------------
// Launch
// ---------------------------------------------------------------------------
extern "C" void kernel_launch(void* const* d_in, const int* in_sizes, int n_in,
                              void* d_out, int out_size, void* d_ws, size_t ws_size,
                              hipStream_t stream) {
  (void)in_sizes; (void)n_in; (void)out_size; (void)ws_size;
  const int*   words   = (const int*)  d_in[0];
  const int*   chars   = (const int*)  d_in[1];
  const float* wl_emb  = (const float*)d_in[2];
  const float* cl_emb  = (const float*)d_in[3];
  const float* c_wih   = (const float*)d_in[4];
  const float* c_whh   = (const float*)d_in[5];
  const float* c_b     = (const float*)d_in[6];
  const float* l0f_wih = (const float*)d_in[7];
  const float* l0f_whh = (const float*)d_in[8];
  const float* l0f_b   = (const float*)d_in[9];
  const float* l0b_wih = (const float*)d_in[10];
  const float* l0b_whh = (const float*)d_in[11];
  const float* l0b_b   = (const float*)d_in[12];
  const float* lf_wih  = (const float*)d_in[13];
  const float* lf_whh  = (const float*)d_in[14];
  const float* lf_b    = (const float*)d_in[15];
  const float* lb_wih  = (const float*)d_in[16];
  const float* lb_whh  = (const float*)d_in[17];
  const float* lb_b    = (const float*)d_in[18];
  const float* Wc      = (const float*)d_in[19];
  const float* bc      = (const float*)d_in[20];
  float* out = (float*)d_out;

  char* base = (char*)d_ws;
  size_t off = 0;
  auto alloc = [&](size_t elems) -> unsigned short* {
    unsigned short* p = (unsigned short*)(base + off);
    off += ((elems * 2) + 255) & ~(size_t)255;
    return p;
  };
  unsigned short* X0 = alloc((size_t)16384 * 320);    // [wl | char-h] bf16
  unsigned short* CE = alloc((size_t)262144 * 64);    // char embeddings
  unsigned short* HA = alloc((size_t)16384 * 1024);   // layer IO ping
  unsigned short* HB = alloc((size_t)16384 * 1024);   // layer IO pong
  unsigned short* PF = alloc((size_t)16384 * 2048);   // pre (fwd), gate-interleaved
  unsigned short* PB = alloc((size_t)16384 * 2048);   // pre (bwd), gate-interleaved
  unsigned short* cwih_b = alloc((size_t)256 * 64);
  unsigned short* cwhh_b = alloc((size_t)256 * 64);
  unsigned short* w0f_ih = alloc((size_t)2048 * 320);
  unsigned short* w0f_hh = alloc((size_t)2048 * 512);
  unsigned short* w0b_ih = alloc((size_t)2048 * 320);
  unsigned short* w0b_hh = alloc((size_t)2048 * 512);
  unsigned short* wf_ih  = alloc((size_t)2 * 2048 * 1024);
  unsigned short* wf_hh  = alloc((size_t)2 * 2048 * 512);
  unsigned short* wb_ih  = alloc((size_t)2 * 2048 * 1024);
  unsigned short* wb_hh  = alloc((size_t)2 * 2048 * 512);

  auto CVT = [&](const float* s, unsigned short* d, int n) {
    k_cvt<<<(n + 255) / 256, 256, 0, stream>>>(s, d, n);
  };
  CVT(c_wih, cwih_b, 256 * 64);
  CVT(c_whh, cwhh_b, 256 * 64);
  CVT(l0f_wih, w0f_ih, 2048 * 320);
  CVT(l0f_whh, w0f_hh, 2048 * 512);
  CVT(l0b_wih, w0b_ih, 2048 * 320);
  CVT(l0b_whh, w0b_hh, 2048 * 512);
  CVT(lf_wih, wf_ih, 2 * 2048 * 1024);
  CVT(lf_whh, wf_hh, 2 * 2048 * 512);
  CVT(lb_wih, wb_ih, 2 * 2048 * 1024);
  CVT(lb_whh, wb_hh, 2 * 2048 * 512);

  k_embed_words<<<16384, 256, 0, stream>>>(words, wl_emb, X0);
  k_embed_chars<<<65536, 256, 0, stream>>>(chars, cl_emb, CE);
  k_char_lstm<<<1024, 128, 0, stream>>>(CE, cwih_b, cwhh_b, c_b, X0);

  // Layer 0: X0(16384x320) -> HA
  k_gemm_pre<<<dim3(128, 16), 256, 0, stream>>>(X0, w0f_ih, l0f_b, PF, 16384, 320);
  k_gemm_pre<<<dim3(128, 16), 256, 0, stream>>>(X0, w0b_ih, l0b_b, PB, 16384, 320);
  k_lstm_rec<<<4, 1024, 0, stream>>>(PF, PB, w0f_hh, w0b_hh, HA);

  // Layer 1: HA(16384x1024) -> HB
  k_gemm_pre<<<dim3(128, 16), 256, 0, stream>>>(HA, wf_ih, lf_b, PF, 16384, 1024);
  k_gemm_pre<<<dim3(128, 16), 256, 0, stream>>>(HA, wb_ih, lb_b, PB, 16384, 1024);
  k_lstm_rec<<<4, 1024, 0, stream>>>(PF, PB, wf_hh, wb_hh, HB);

  // Layer 2: HB -> HA
  k_gemm_pre<<<dim3(128, 16), 256, 0, stream>>>(HB, wf_ih + 2048 * 1024, lf_b + 2048, PF, 16384, 1024);
  k_gemm_pre<<<dim3(128, 16), 256, 0, stream>>>(HB, wb_ih + 2048 * 1024, lb_b + 2048, PB, 16384, 1024);
  k_lstm_rec<<<4, 1024, 0, stream>>>(PF, PB, wf_hh + 2048 * 512, wb_hh + 2048 * 512, HA);

  k_classifier<<<1, 192, 0, stream>>>(HA, Wc, bc, out);
}
